// LinearAttentionHead_85925115723780
// MI455X (gfx1250) — matrix operand order repaired and checked by reading, not compiled, not run on hardware
//
#include <hip/hip_runtime.h>

typedef _Float16 v16h __attribute__((ext_vector_type(16)));
typedef _Float16 v8h  __attribute__((ext_vector_type(8)));
typedef float    v8f  __attribute__((ext_vector_type(8)));

#define WMMA(a, b, c) __builtin_amdgcn_wmma_f32_16x16x32_f16( \
    false, (a), false, (b), (short)0, (c), false, false)

static __device__ __forceinline__ v16h cat16(v8h lo, v8h hi) {
  v16h r;
#pragma unroll
  for (int i = 0; i < 8; ++i) { r[i] = lo[i]; r[i + 8] = hi[i]; }
  return r;
}

// Async raw 16-byte copy global -> LDS (CDNA5 GLOBAL_LOAD_ASYNC_TO_LDS_B128,
// tracked by ASYNCcnt). lds generic pointer's low 32 bits are the LDS offset.
static __device__ __forceinline__ void async_cp16(void* lds, const void* g) {
  unsigned lds_off = (unsigned)(uintptr_t)lds;
  unsigned long long ga = (unsigned long long)(uintptr_t)g;
  asm volatile("global_load_async_to_lds_b128 %0, %1, off"
               :: "v"(lds_off), "v"(ga) : "memory");
}
static __device__ __forceinline__ void wait_async0() {
  asm volatile("s_wait_asynccnt 0" ::: "memory");
}

// ---------------------------------------------------------------------------
// Stage 1: KpT[b][k][d] = sum_n K[b][n][d]*E[n][k]   (stored k-major, f16)
//          VpT[b][d][k] = sum_n V[b][n][d]*F[n][k]   (stored d-major, f16)
// grid: 32 batches * 8 k-tiles(32 wide), 256 threads (8 waves)
// waves 0-3: Kp (64d x 32k), waves 4-7: Vp (32k x 64d)
// ---------------------------------------------------------------------------
__global__ __launch_bounds__(256) void proj_kernel(
    const float* __restrict__ Kg, const float* __restrict__ Vg,
    const float* __restrict__ Eg, const float* __restrict__ Fg,
    _Float16* __restrict__ KpT, _Float16* __restrict__ VpT) {
  const int b   = blockIdx.x >> 3;
  const int k0  = (blockIdx.x & 7) * 32;
  const int tid = threadIdx.x;
  const int lane = tid & 31;
  const int wave = tid >> 5;
  const int hl   = lane >> 4;   // wave half
  const int l16  = lane & 15;

  // transposed chunk staging: s*[row][n], n = position within 32-chunk
  __shared__ _Float16 sK[64][48];   // [d][n]
  __shared__ _Float16 sV[64][48];   // [d][n]
  __shared__ _Float16 sE[32][48];   // [k][n]
  __shared__ _Float16 sF[32][48];   // [k][n]

  const bool doKp  = wave < 4;
  const int  Mbase = doKp ? wave * 16 : (((wave - 4) >> 1) * 16);
  const int  ctb   = doKp ? 0 : (((wave - 4) & 1) * 2);  // Vp: d col-tile pair

  v8f acc0 = {}, acc1 = {};

  for (int n0 = 0; n0 < 4096; n0 += 32) {
    __syncthreads();
    // K,V chunks: 32 rows x 64 cols (512 float4 each -> 2 per thread)
#pragma unroll
    for (int r = 0; r < 2; ++r) {
      int i  = tid + r * 256;
      int n  = i >> 4;
      int d4 = (i & 15) * 4;
      size_t g = ((size_t)b * 4096 + n0 + n) * 64 + d4;
      float4 kv = *(const float4*)(Kg + g);
      sK[d4 + 0][n] = (_Float16)kv.x; sK[d4 + 1][n] = (_Float16)kv.y;
      sK[d4 + 2][n] = (_Float16)kv.z; sK[d4 + 3][n] = (_Float16)kv.w;
      float4 vv = *(const float4*)(Vg + g);
      sV[d4 + 0][n] = (_Float16)vv.x; sV[d4 + 1][n] = (_Float16)vv.y;
      sV[d4 + 2][n] = (_Float16)vv.z; sV[d4 + 3][n] = (_Float16)vv.w;
    }
    // E,F chunks: 32 rows x 32 cols (256 float4 each -> 1 per thread)
    {
      int n  = tid >> 3;
      int c4 = (tid & 7) * 4;
      size_t g = (size_t)(n0 + n) * 256 + k0 + c4;
      float4 ev = *(const float4*)(Eg + g);
      sE[c4 + 0][n] = (_Float16)ev.x; sE[c4 + 1][n] = (_Float16)ev.y;
      sE[c4 + 2][n] = (_Float16)ev.z; sE[c4 + 3][n] = (_Float16)ev.w;
      float4 fv = *(const float4*)(Fg + g);
      sF[c4 + 0][n] = (_Float16)fv.x; sF[c4 + 1][n] = (_Float16)fv.y;
      sF[c4 + 2][n] = (_Float16)fv.z; sF[c4 + 3][n] = (_Float16)fv.w;
    }
    __syncthreads();

    // A fragment: 16x32, row per lane (hl=0: K 0-7/16-23, hl=1: 8-15/24-31)
    v8h alo, ahi;
    if (doKp) {
      alo = *(const v8h*)&sK[Mbase + l16][hl * 8];
      ahi = *(const v8h*)&sK[Mbase + l16][16 + hl * 8];
    } else {
      alo = *(const v8h*)&sF[Mbase + l16][hl * 8];
      ahi = *(const v8h*)&sF[Mbase + l16][16 + hl * 8];
    }
    v16h a = cat16(alo, ahi);

    // B fragments: 32x16, col per lane (hl=0: K 0-7/16-23, hl=1: 8-15/24-31)
    v16h b0, b1;
    if (doKp) {
      b0 = cat16(*(const v8h*)&sE[l16][hl * 8],
                 *(const v8h*)&sE[l16][16 + hl * 8]);
      b1 = cat16(*(const v8h*)&sE[16 + l16][hl * 8],
                 *(const v8h*)&sE[16 + l16][16 + hl * 8]);
    } else {
      b0 = cat16(*(const v8h*)&sV[ctb * 16 + l16][hl * 8],
                 *(const v8h*)&sV[ctb * 16 + l16][16 + hl * 8]);
      b1 = cat16(*(const v8h*)&sV[(ctb + 1) * 16 + l16][hl * 8],
                 *(const v8h*)&sV[(ctb + 1) * 16 + l16][16 + hl * 8]);
    }
    acc0 = WMMA(a, b0, acc0);
    acc1 = WMMA(a, b1, acc1);
  }

  // D layout: VGPR r, lane -> row M = r + hl*8, col N = l16
  if (doKp) {
#pragma unroll
    for (int r = 0; r < 8; ++r) {
      int didx = Mbase + r + hl * 8;
      KpT[((size_t)b * 256 + k0 + l16) * 64 + didx]      = (_Float16)acc0[r];
      KpT[((size_t)b * 256 + k0 + 16 + l16) * 64 + didx] = (_Float16)acc1[r];
    }
  } else {
#pragma unroll
    for (int r = 0; r < 8; ++r) {
      int kidx = k0 + Mbase + r + hl * 8;
      VpT[((size_t)b * 64 + ctb * 16 + l16) * 256 + kidx]       = (_Float16)acc0[r];
      VpT[((size_t)b * 64 + (ctb + 1) * 16 + l16) * 256 + kidx] = (_Float16)acc1[r];
    }
  }
}

// ---------------------------------------------------------------------------
// Stage 2: scores = (Q/8) * Kp  -> softmax over k=256 -> out = P * Vp
// grid: 32 batches * 64 row-tiles(64 rows), 128 threads (4 waves, 16 rows each)
// Kp[b] / Vp[b] staged into LDS once per block via async-LDS loads.
// ---------------------------------------------------------------------------
__global__ __launch_bounds__(128) void attn_kernel(
    const float* __restrict__ Qg,
    const _Float16* __restrict__ KpT, const _Float16* __restrict__ VpT,
    float* __restrict__ Out) {
  const int b   = blockIdx.x >> 6;
  const int n0  = (blockIdx.x & 63) * 64;
  const int tid = threadIdx.x;
  const int lane = tid & 31;
  const int wave = tid >> 5;
  const int hl   = lane >> 4;
  const int l16  = lane & 15;
  const int Mb   = wave * 16;

  __shared__ _Float16 sQ[64][72];     // row-major Q tile (f16, pre-scaled)
  __shared__ _Float16 sP[64][264];    // row-major P (probabilities)
  __shared__ _Float16 sKp[256][72];   // Kp[b]  [k][d], padded rows
  __shared__ _Float16 sVp[64][264];   // Vp[b]  [d][k], padded rows

  // -- async-stage Kp[b] (256x64 f16) and Vp[b] (64x256 f16) into LDS --
  {
    const _Float16* kb = KpT + (size_t)b * 256 * 64;
#pragma unroll
    for (int i = 0; i < 16; ++i) {
      int chunk = tid + i * 128;          // 2048 x 16B
      int r = chunk >> 3, cg = (chunk & 7) * 8;
      async_cp16(&sKp[r][cg], kb + r * 64 + cg);
    }
    const _Float16* vb = VpT + (size_t)b * 64 * 256;
#pragma unroll
    for (int i = 0; i < 16; ++i) {
      int chunk = tid + i * 128;          // 2048 x 16B
      int r = chunk >> 5, cg = (chunk & 31) * 8;
      async_cp16(&sVp[r][cg], vb + r * 256 + cg);
    }
  }

  // -- stage Q tile (f32 -> f16, scaled by 1/sqrt(64)) while copies fly --
#pragma unroll
  for (int r = 0; r < 8; ++r) {
    int i   = tid + r * 128;
    int row = i >> 4;
    int d4  = (i & 15) * 4;
    float4 qv = *(const float4*)(Qg + ((size_t)b * 4096 + n0 + row) * 64 + d4);
    sQ[row][d4 + 0] = (_Float16)(qv.x * 0.125f);
    sQ[row][d4 + 1] = (_Float16)(qv.y * 0.125f);
    sQ[row][d4 + 2] = (_Float16)(qv.z * 0.125f);
    sQ[row][d4 + 3] = (_Float16)(qv.w * 0.125f);
  }
  wait_async0();
  __syncthreads();

  // Q A-fragments: 2 chunks over d=64
  v16h aq[2];
#pragma unroll
  for (int c = 0; c < 2; ++c) {
    v8h lo = *(const v8h*)&sQ[Mb + l16][c * 32 + hl * 8];
    v8h hi = *(const v8h*)&sQ[Mb + l16][c * 32 + 16 + hl * 8];
    aq[c] = cat16(lo, hi);
  }

  // GEMM1: scores 16x256 per wave (16 accumulators)
  v8f acc[16];
#pragma unroll
  for (int ct = 0; ct < 16; ++ct) acc[ct] = (v8f){};
#pragma unroll
  for (int c = 0; c < 2; ++c) {
#pragma unroll
    for (int ct = 0; ct < 16; ++ct) {
      v16h bf = cat16(*(const v8h*)&sKp[ct * 16 + l16][c * 32 + hl * 8],
                      *(const v8h*)&sKp[ct * 16 + l16][c * 32 + 16 + hl * 8]);
      acc[ct] = WMMA(aq[c], bf, acc[ct]);
    }
  }

  // softmax over 256 cols: each row lives in 16 lanes of a half-wave
  v8f mx = acc[0];
#pragma unroll
  for (int ct = 1; ct < 16; ++ct)
#pragma unroll
    for (int e = 0; e < 8; ++e) mx[e] = fmaxf(mx[e], acc[ct][e]);
#pragma unroll
  for (int off = 1; off < 16; off <<= 1)
#pragma unroll
    for (int e = 0; e < 8; ++e) mx[e] = fmaxf(mx[e], __shfl_xor(mx[e], off, 32));

  v8f sum = {};
#pragma unroll
  for (int ct = 0; ct < 16; ++ct)
#pragma unroll
    for (int e = 0; e < 8; ++e) {
      float p = __expf(acc[ct][e] - mx[e]);
      acc[ct][e] = p;
      sum[e] += p;
    }
#pragma unroll
  for (int off = 1; off < 16; off <<= 1)
#pragma unroll
    for (int e = 0; e < 8; ++e) sum[e] += __shfl_xor(sum[e], off, 32);

  v8f inv;
#pragma unroll
  for (int e = 0; e < 8; ++e) inv[e] = 1.0f / sum[e];

  // D-layout -> row-major LDS (transpose for A-fragments of GEMM2)
#pragma unroll
  for (int ct = 0; ct < 16; ++ct)
#pragma unroll
    for (int e = 0; e < 8; ++e)
      sP[Mb + e + hl * 8][ct * 16 + l16] = (_Float16)(acc[ct][e] * inv[e]);
  __syncthreads();

  // GEMM2: out 16x64 per wave, K = 256 in 8 chunks
  v8f oc[4];
#pragma unroll
  for (int dt = 0; dt < 4; ++dt) oc[dt] = (v8f){};
#pragma unroll
  for (int kc = 0; kc < 8; ++kc) {
    v8h lo = *(const v8h*)&sP[Mb + l16][kc * 32 + hl * 8];
    v8h hi = *(const v8h*)&sP[Mb + l16][kc * 32 + 16 + hl * 8];
    v16h ap = cat16(lo, hi);
#pragma unroll
    for (int dt = 0; dt < 4; ++dt) {
      v16h bf = cat16(*(const v8h*)&sVp[dt * 16 + l16][kc * 32 + hl * 8],
                      *(const v8h*)&sVp[dt * 16 + l16][kc * 32 + 16 + hl * 8]);
      oc[dt] = WMMA(ap, bf, oc[dt]);
    }
  }

  // store f32 output, coalesced over l16
#pragma unroll
  for (int dt = 0; dt < 4; ++dt)
#pragma unroll
    for (int e = 0; e < 8; ++e)
      Out[((size_t)b * 4096 + n0 + Mb + e + hl * 8) * 64 + dt * 16 + l16] =
          oc[dt][e];
}

extern "C" void kernel_launch(void* const* d_in, const int* in_sizes, int n_in,
                              void* d_out, int out_size, void* d_ws,
                              size_t ws_size, hipStream_t stream) {
  (void)in_sizes; (void)n_in; (void)out_size; (void)ws_size;
  const float* Q = (const float*)d_in[0];
  const float* K = (const float*)d_in[1];
  const float* V = (const float*)d_in[2];
  const float* E = (const float*)d_in[3];
  const float* F = (const float*)d_in[4];
  float* Out = (float*)d_out;

  _Float16* KpT = (_Float16*)d_ws;                     // [32][256][64]
  _Float16* VpT = KpT + (size_t)32 * 256 * 64;         // [32][64][256]

  proj_kernel<<<32 * 8, 256, 0, stream>>>(K, V, E, F, KpT, VpT);
  attn_kernel<<<32 * 64, 128, 0, stream>>>(Q, KpT, VpT, Out);
}
